// PaiNN_34394098106848
// MI455X (gfx1250) — compile-verified, matrix operand order checked by reading, not run on hardware
//
#include <hip/hip_runtime.h>

// ---------------- problem constants (match reference) ----------------
#define F_DIM   128
#define F3      384
#define F2      256
#define N_RBF   20
#define CUTOFF_R 5.0f
#define L_ITERS 3
#define EPS_N   1e-8f

typedef __attribute__((ext_vector_type(16))) _Float16 v16h;
typedef __attribute__((ext_vector_type(8)))  float    v8f;

#define LDS_AS __attribute__((address_space(3)))
#define GLB_AS __attribute__((address_space(1)))

// exact pointee type expected by the async-load builtin (per hipcc diagnostic:
// "int __attribute__((vector_size(16))) __device__ *")
typedef int v4i_vs __attribute__((vector_size(16)));

#if defined(__gfx1250__) && __has_builtin(__builtin_amdgcn_global_load_async_to_lds_b128)
#define HAVE_ASYNC_LDS 1
#endif

// 16-byte global -> LDS copy.
//  Async path: CDNA5 GLOBAL_LOAD_ASYNC_TO_LDS_B128 (tracked by ASYNCcnt, no VGPR data).
//  as3 pointers are 32-bit wave-relative LDS offsets; generic LDS addresses carry the
//  offset in the low 32 bits, so the integer-truncation cast yields the right value.
__device__ __forceinline__ void cp16_g2l(float* dst_lds, const float* src_glb)
{
#if defined(HAVE_ASYNC_LDS)
    __builtin_amdgcn_global_load_async_to_lds_b128(
        (GLB_AS v4i_vs*)(uintptr_t)src_glb,
        (LDS_AS v4i_vs*)(unsigned int)(uintptr_t)dst_lds,
        /*imm offset*/0, /*cpol*/0);
#else
    *(float4*)dst_lds = *(const float4*)src_glb;
#endif
}

__device__ __forceinline__ void wait_async_zero()
{
#if defined(HAVE_ASYNC_LDS)
#if __has_builtin(__builtin_amdgcn_s_wait_asynccnt)
    __builtin_amdgcn_s_wait_asynccnt(0);
#else
    asm volatile("s_wait_asynccnt 0x0" ::: "memory");
#endif
#endif
}

__device__ __forceinline__ float silu_f(float v)
{
    return v * __builtin_amdgcn_rcpf(1.0f + __expf(-v));
}

// =====================================================================
// WMMA GEMM:  C[M,Nc] = act(A[M,K] @ W[Nc,K]^T + bias)
//   Block = 256 threads (8 waves). Output tile 128x64, K-step 32.
//   Tiles staged to LDS in f32 (async, double-buffered), converted to
//   f16 at fragment build. Each wave: 4 m-tiles x 1 n-tile = 4 WMMAs
//   per K-step sharing one B fragment.  act: 0 = identity, 1 = SiLU.
//   Requires: M % 128 == 0, Nc % 64 == 0, K % 32 == 0 (true here).
// =====================================================================
#define TBM 128
#define TBN 64
#define TBK 32

__global__ __launch_bounds__(256)
void gemm_wmma_f16(const float* __restrict__ A, const float* __restrict__ W,
                   const float* __restrict__ bias, float* __restrict__ C,
                   int M, int K, int Nc, int act)
{
    __shared__ __align__(16) float As[2][TBM][TBK];   // 32 KB
    __shared__ __align__(16) float Ws[2][TBN][TBK];   // 16 KB  (W row-major [out,in] == B^T)

    const int tid  = threadIdx.x;
    const int wave = tid >> 5;
    const int lane = tid & 31;
    const int hi   = lane >> 4;
    const int lm   = lane & 15;

    const int block_m = (int)blockIdx.x * TBM;
    const int block_n = (int)blockIdx.y * TBN;

    const int mtb = (wave >> 2) * 4;   // 0 or 4: base m-tile (4 per wave)
    const int nt  = wave & 3;          // n-tile

    v8f acc[4] = {v8f{}, v8f{}, v8f{}, v8f{}};

    auto stage = [&](int buf, int k0) {
        // A tile 128x32 f32: 1024 x 16B, 4 per thread
        #pragma unroll
        for (int it = 0; it < 4; ++it) {
            int idx = tid + it * 256;
            int r = idx >> 3;
            int c = (idx & 7) << 2;
            cp16_g2l(&As[buf][r][c], &A[(size_t)(block_m + r) * K + k0 + c]);
        }
        // W tile 64x32 f32: 512 x 16B, 2 per thread
        #pragma unroll
        for (int it = 0; it < 2; ++it) {
            int idx = tid + it * 256;
            int r = idx >> 3;
            int c = (idx & 7) << 2;
            cp16_g2l(&Ws[buf][r][c], &W[(size_t)(block_n + r) * K + k0 + c]);
        }
    };

    stage(0, 0);
    int buf = 0;
    for (int k0 = 0; k0 < K; k0 += TBK, buf ^= 1) {
        wait_async_zero();          // previous issue into `buf` landed in LDS
        __syncthreads();            // ... and visible to all waves
        if (k0 + TBK < K) stage(buf ^ 1, k0 + TBK);   // overlap next stage with WMMA

        // ISA 16-bit A-operand layout: lane = row + 16*hi,
        //   element e -> k = 16*(e/8) + 8*hi + (e%8).  B mirrors with n as row.
        v16h bfrag;
        #pragma unroll
        for (int e = 0; e < 16; ++e) {
            int kk = ((e >> 3) << 4) + (hi << 3) + (e & 7);
            bfrag[e] = (_Float16)Ws[buf][nt * 16 + lm][kk];
        }
        #pragma unroll
        for (int t = 0; t < 4; ++t) {
            v16h afrag;
            #pragma unroll
            for (int e = 0; e < 16; ++e) {
                int kk = ((e >> 3) << 4) + (hi << 3) + (e & 7);
                afrag[e] = (_Float16)As[buf][(mtb + t) * 16 + lm][kk];
            }
            acc[t] = __builtin_amdgcn_wmma_f32_16x16x32_f16(false, afrag, false, bfrag,
                                                            (short)0, acc[t], false, false);
        }
        __syncthreads();            // everyone done reading `buf` before it is re-staged
    }

    // C/D layout: lane(0..15)=N, VGPR r -> M = r (+8 for lanes 16..31)
    const int n = block_n + nt * 16 + lm;
    float bv = (bias != nullptr) ? bias[n] : 0.0f;
    #pragma unroll
    for (int t = 0; t < 4; ++t) {
        #pragma unroll
        for (int r = 0; r < 8; ++r) {
            int m = block_m + (mtb + t) * 16 + r + (hi << 3);
            float v = acc[t][r] + bv;
            if (act) v = silu_f(v);
            if (m < M && n < Nc) C[(size_t)m * Nc + n] = v;
        }
    }
}

// =====================================================================
// Init: q = emb[x], mu = 0
// =====================================================================
__global__ void init_qmu(const float* __restrict__ emb, const int* __restrict__ x,
                         float* __restrict__ q, float* __restrict__ mu, int n_atoms)
{
    int idx = blockIdx.x * blockDim.x + threadIdx.x;
    if (idx >= n_atoms * F_DIM) return;
    int n = idx >> 7, f = idx & 127;
    q[idx] = emb[(size_t)x[n] * F_DIM + f];
    size_t b = (size_t)n * F3 + f;
    mu[b] = 0.0f; mu[b + F_DIM] = 0.0f; mu[b + 2 * F_DIM] = 0.0f;
}

__global__ void zero_buf(float* __restrict__ p, int n)
{
    int idx = blockIdx.x * blockDim.x + threadIdx.x;
    if (idx < n) p[idx] = 0.0f;
}

// =====================================================================
// Edge message kernel (per layer).  Recomputes RBF filter Wij on the fly
// (filt_W staged in LDS), gathers h[j], mu_old[j], scatter-adds q/mu.
// 256 threads = 2 edges x 128 feature lanes; grid-stride over edge pairs.
// Prefetches next iteration's gather rows (global_prefetch_b8).
// =====================================================================
__global__ __launch_bounds__(256)
void edge_message(const float* __restrict__ pos,
                  const float* __restrict__ filt_W, const float* __restrict__ filt_b,
                  const float* __restrict__ h, const float* __restrict__ mu_old,
                  const int* __restrict__ idx_i, const int* __restrict__ idx_j,
                  float* __restrict__ q, float* __restrict__ mu, int n_edges)
{
    __shared__ float sW[F3 * N_RBF];     // 30720 B
    __shared__ float sb[F3];
    __shared__ float sphi[2][N_RBF];

    for (int i = threadIdx.x; i < F3 * N_RBF; i += 256) sW[i] = filt_W[i];
    for (int i = threadIdx.x; i < F3;         i += 256) sb[i] = filt_b[i];
    __syncthreads();

    const int sub = threadIdx.x >> 7;          // which edge of the pair
    const int f   = threadIdx.x & 127;
    const float width = CUTOFF_R / (float)(N_RBF - 1);
    const float coeff = -0.5f / (width * width);
    const float pi_over_rc = 3.14159265358979f / CUTOFF_R;
    const long long estride = (long long)gridDim.x * 2;

    for (long long pair = blockIdx.x; pair * 2 < (long long)n_edges; pair += gridDim.x) {
        long long e = pair * 2 + sub;
        bool valid = (e < (long long)n_edges);
        int ai = 0, aj = 0;
        float dirx = 0.f, diry = 0.f, dirz = 0.f, fcut = 0.f;

        __syncthreads();   // sphi reuse across iterations
        if (valid) {
            ai = idx_i[e]; aj = idx_j[e];
            // hide next iteration's gather latency behind the filter math
            long long en = e + estride;
            if (en < (long long)n_edges) {
                int jn = idx_j[en];
                __builtin_prefetch(&h[(size_t)jn * F3 + f], 0, 1);
                __builtin_prefetch(&mu_old[(size_t)jn * F3 + f], 0, 1);
            }
            float rx = pos[(size_t)aj * 3 + 0] - pos[(size_t)ai * 3 + 0];
            float ry = pos[(size_t)aj * 3 + 1] - pos[(size_t)ai * 3 + 1];
            float rz = pos[(size_t)aj * 3 + 2] - pos[(size_t)ai * 3 + 2];
            float d  = sqrtf(rx * rx + ry * ry + rz * rz);
            float inv = 1.0f / fmaxf(d, 1e-12f);
            dirx = rx * inv; diry = ry * inv; dirz = rz * inv;
            fcut = (d < CUTOFF_R) ? 0.5f * (__cosf(d * pi_over_rc) + 1.0f) : 0.0f;
            if (f < N_RBF) {
                float del = d - (float)f * width;
                sphi[sub][f] = __expf(coeff * del * del);
            }
        }
        __syncthreads();

        if (valid) {
            // Wij rows f, f+128, f+256 : dot(filt_W, phi) + b, times fcut
            float w0 = sb[f], w1 = sb[f + F_DIM], w2 = sb[f + 2 * F_DIM];
            #pragma unroll
            for (int r = 0; r < N_RBF; ++r) {
                float p = sphi[sub][r];
                w0 += sW[f * N_RBF + r] * p;
                w1 += sW[(f + F_DIM) * N_RBF + r] * p;
                w2 += sW[(f + 2 * F_DIM) * N_RBF + r] * p;
            }
            w0 *= fcut; w1 *= fcut; w2 *= fcut;

            const float* hj = h + (size_t)aj * F3;
            float xq = w0 * hj[f];                 // dq
            float xr = w1 * hj[f + F_DIM];         // dmuR
            float xm = w2 * hj[f + 2 * F_DIM];     // dmumu

            const float* mj = mu_old + (size_t)aj * F3;
            float m0 = mj[f], m1 = mj[F_DIM + f], m2 = mj[2 * F_DIM + f];

            atomicAdd(&q[(size_t)ai * F_DIM + f], xq);
            float* mi = mu + (size_t)ai * F3;
            atomicAdd(&mi[f],             xr * dirx + xm * m0);
            atomicAdd(&mi[F_DIM + f],     xr * diry + xm * m1);
            atomicAdd(&mi[2 * F_DIM + f], xr * dirz + xm * m2);
        }
    }
}

// =====================================================================
// Mixing prep: ctx = [q, |mu_V|], muVW = sum_axis(mu_V * mu_W)
// mu_mix is [N, 3, 256] (first 128 = V, last 128 = W)
// =====================================================================
__global__ void mix_prep(const float* __restrict__ q, const float* __restrict__ mu_mix,
                         float* __restrict__ ctx, float* __restrict__ muVW, int n_atoms)
{
    int idx = blockIdx.x * blockDim.x + threadIdx.x;
    if (idx >= n_atoms * F_DIM) return;
    int n = idx >> 7, f = idx & 127;
    const float* mm = mu_mix + (size_t)n * 3 * F2;
    float v0 = mm[0 * F2 + f], v1 = mm[1 * F2 + f], v2 = mm[2 * F2 + f];
    float w0 = mm[0 * F2 + F_DIM + f], w1 = mm[1 * F2 + F_DIM + f], w2 = mm[2 * F2 + F_DIM + f];
    float vn = sqrtf(v0 * v0 + v1 * v1 + v2 * v2 + EPS_N);
    ctx[(size_t)n * F2 + f]          = q[idx];
    ctx[(size_t)n * F2 + F_DIM + f]  = vn;
    muVW[idx] = v0 * w0 + v1 * w1 + v2 * w2;
}

// =====================================================================
// Mixing update: q += dq + dqmu*muVW ; mu += dmu * mu_W
// g = [N,384] (dq | dmu | dqmu)
// =====================================================================
__global__ void mix_update(const float* __restrict__ g, const float* __restrict__ mu_mix,
                           const float* __restrict__ muVW,
                           float* __restrict__ q, float* __restrict__ mu, int n_atoms)
{
    int idx = blockIdx.x * blockDim.x + threadIdx.x;
    if (idx >= n_atoms * F_DIM) return;
    int n = idx >> 7, f = idx & 127;
    const float* gg = g + (size_t)n * F3;
    q[idx] += gg[f] + gg[2 * F_DIM + f] * muVW[idx];
    float dmu = gg[F_DIM + f];
    const float* mm = mu_mix + (size_t)n * 3 * F2;
    float* mi = mu + (size_t)n * F3;
    mi[f]             += dmu * mm[0 * F2 + F_DIM + f];
    mi[F_DIM + f]     += dmu * mm[1 * F2 + F_DIM + f];
    mi[2 * F_DIM + f] += dmu * mm[2 * F2 + F_DIM + f];
}

// =====================================================================
// Readout: out[batch[n], f] += q[n, f]
// =====================================================================
__global__ void readout(const float* __restrict__ q, const int* __restrict__ batch,
                        float* __restrict__ out, int n_atoms)
{
    int idx = blockIdx.x * blockDim.x + threadIdx.x;
    if (idx >= n_atoms * F_DIM) return;
    int n = idx >> 7, f = idx & 127;
    atomicAdd(&out[(size_t)batch[n] * F_DIM + f], q[idx]);
}

// =====================================================================
extern "C" void kernel_launch(void* const* d_in, const int* in_sizes, int n_in,
                              void* d_out, int out_size, void* d_ws, size_t ws_size,
                              hipStream_t stream)
{
    const float* pos      = (const float*)d_in[0];
    const float* emb      = (const float*)d_in[1];
    const float* filt_W   = (const float*)d_in[2];
    const float* filt_b   = (const float*)d_in[3];
    const float* inter_W1 = (const float*)d_in[4];
    const float* inter_b1 = (const float*)d_in[5];
    const float* inter_W2 = (const float*)d_in[6];
    const float* inter_b2 = (const float*)d_in[7];
    const float* mix_W1   = (const float*)d_in[8];
    const float* mix_b1   = (const float*)d_in[9];
    const float* mix_W2   = (const float*)d_in[10];
    const float* mix_b2   = (const float*)d_in[11];
    const float* mix_Wmu  = (const float*)d_in[12];
    const int*   x        = (const int*)d_in[13];
    const int*   idx_i    = (const int*)d_in[14];
    const int*   idx_j    = (const int*)d_in[15];
    const int*   batch    = (const int*)d_in[16];

    const int n_atoms = in_sizes[13];
    const int n_edges = in_sizes[14];
    float* out = (float*)d_out;

    // ---- workspace layout (all f32) ----
    float* p = (float*)d_ws;
    float* q      = p; p += (size_t)n_atoms * F_DIM;       // [N,128]
    float* mu     = p; p += (size_t)n_atoms * F3;          // [N,3,128]
    float* mu_old = p; p += (size_t)n_atoms * F3;          // snapshot for gather
    float* h1     = p; p += (size_t)n_atoms * F_DIM;       // [N,128] hidden
    float* h      = p; p += (size_t)n_atoms * F3;          // [N,384] (h / g)
    float* mu_mix = p; p += (size_t)n_atoms * 3 * F2;      // [N,3,256]
    float* ctx    = p; p += (size_t)n_atoms * F2;          // [N,256]
    float* muVW   = p; p += (size_t)n_atoms * F_DIM;       // [N,128]

    const int elems = n_atoms * F_DIM;
    const int tpb = 256;

    auto gemm = [&](const float* A, const float* W, const float* b, float* C,
                    int M, int K, int Nc, int act) {
        dim3 grid((M + TBM - 1) / TBM, (Nc + TBN - 1) / TBN);
        gemm_wmma_f16<<<grid, 256, 0, stream>>>(A, W, b, C, M, K, Nc, act);
    };

    // init
    zero_buf<<<(out_size + tpb - 1) / tpb, tpb, 0, stream>>>(out, out_size);
    init_qmu<<<(elems + tpb - 1) / tpb, tpb, 0, stream>>>(emb, x, q, mu, n_atoms);

    for (int l = 0; l < L_ITERS; ++l) {
        // --- interaction block ---
        gemm(q,  inter_W1 + (size_t)l * F_DIM * F_DIM, inter_b1 + (size_t)l * F_DIM,
             h1, n_atoms, F_DIM, F_DIM, /*silu*/1);
        gemm(h1, inter_W2 + (size_t)l * F3 * F_DIM,    inter_b2 + (size_t)l * F3,
             h,  n_atoms, F_DIM, F3, 0);

        // snapshot mu for gather semantics (reference reads old mu[idx_j])
        (void)hipMemcpyAsync(mu_old, mu, (size_t)n_atoms * F3 * sizeof(float),
                             hipMemcpyDeviceToDevice, stream);

        edge_message<<<8192, 256, 0, stream>>>(pos, filt_W, filt_b, h, mu_old,
                                               idx_i, idx_j, q, mu, n_edges);

        // --- mixing block ---
        gemm(mu, mix_Wmu + (size_t)l * F2 * F_DIM, nullptr,
             mu_mix, 3 * n_atoms, F_DIM, F2, 0);

        mix_prep<<<(elems + tpb - 1) / tpb, tpb, 0, stream>>>(q, mu_mix, ctx, muVW, n_atoms);

        gemm(ctx, mix_W1 + (size_t)l * F_DIM * F2, mix_b1 + (size_t)l * F_DIM,
             h1, n_atoms, F2, F_DIM, /*silu*/1);
        gemm(h1,  mix_W2 + (size_t)l * F3 * F_DIM, mix_b2 + (size_t)l * F3,
             h,  n_atoms, F_DIM, F3, 0);

        mix_update<<<(elems + tpb - 1) / tpb, tpb, 0, stream>>>(h, mu_mix, muVW, q, mu, n_atoms);
    }

    readout<<<(elems + tpb - 1) / tpb, tpb, 0, stream>>>(q, batch, out, n_atoms);
}
